// MultiHeadAttentionV2_8220567404937
// MI455X (gfx1250) — matrix operand order repaired and checked by reading, not compiled, not run on hardware
//
#include <hip/hip_runtime.h>
#include <stdint.h>

#define BATCH  2
#define SEQ    2048
#define DMODEL 1024
#define NHEAD  16
#define HDIM   64
#define MTOT   (BATCH * SEQ)   // 4096
#define NQKV   (3 * DMODEL)    // 3072

typedef __bf16 v16bf __attribute__((ext_vector_type(16)));
typedef float  v8f   __attribute__((ext_vector_type(8)));
typedef unsigned int u32x4 __attribute__((ext_vector_type(4)));
typedef int          i32x4 __attribute__((ext_vector_type(4)));
typedef int          i32x8 __attribute__((ext_vector_type(8)));

union Frag16 { uint4 q[2]; v16bf v; };

__device__ __forceinline__ unsigned short f32_to_bf16(float f) {
  unsigned int u = __float_as_uint(f);
  u += 0x7FFFu + ((u >> 16) & 1u);          // round-to-nearest-even
  return (unsigned short)(u >> 16);
}

__device__ __forceinline__ v8f wmma_bf16(Frag16 a, Frag16 b, v8f c) {
  return __builtin_amdgcn_wmma_f32_16x16x32_bf16(
      false, a.v, false, b.v, (short)0, c, false, false);
}

// ---------------------------------------------------------------------------
// Tensor Data Mover 2D bf16 tile descriptor (CDNA5 ISA).
// Group1 + static group0 bits are precomputed; per-issue we patch only the
// LDS offset and the 57-bit global byte address. pad_en inserts a 4-DWORD pad
// after every 16 DWORDs => 40-element padded LDS rows (conflict-free frags).
// ---------------------------------------------------------------------------
struct TdmDesc { i32x8 g1; };

__device__ __forceinline__ TdmDesc tdm_make_2d(unsigned tensor_d0, unsigned tensor_d1,
                                               unsigned stride0,
                                               unsigned tile0, unsigned tile1,
                                               bool pad_en) {
  TdmDesc d;
  unsigned d0 = (1u << 16);                                 // data_size = 2 bytes
  if (pad_en) d0 |= (1u << 20) | (3u << 22) | (3u << 25);   // pad: every 16 DW, +4 DW
  d.g1[0] = (int)d0;
  d.g1[1] = (int)((tensor_d0 & 0xFFFFu) << 16);             // tensor_dim0[15:0]
  d.g1[2] = (int)(((tensor_d0 >> 16) & 0xFFFFu) | ((tensor_d1 & 0xFFFFu) << 16));
  d.g1[3] = (int)(((tensor_d1 >> 16) & 0xFFFFu) | ((tile0 & 0xFFFFu) << 16));
  d.g1[4] = (int)(tile1 & 0xFFFFu);                         // tile_dim1, tile_dim2=0
  d.g1[5] = (int)stride0;                                   // dim0 stride (elems)
  d.g1[6] = 0;
  d.g1[7] = 0;
  return d;
}

__device__ __forceinline__ void tdm_issue(const TdmDesc& d, unsigned lds_off,
                                          const void* gptr) {
  unsigned long long ga = (unsigned long long)(uintptr_t)gptr;
  u32x4 g0;
  g0.x = 0x1u;                                              // count=1 (valid user D#)
  g0.y = lds_off;                                           // LDS byte address
  g0.z = (unsigned)(ga & 0xFFFFFFFFu);                      // global_addr[31:0]
  g0.w = (unsigned)((ga >> 32) & 0x01FFFFFFu) | (2u << 30); // addr[56:32] | type=2
  i32x4 z = {0, 0, 0, 0};
#if __clang_major__ >= 23
  i32x8 z8 = {0, 0, 0, 0, 0, 0, 0, 0};
  __builtin_amdgcn_tensor_load_to_lds(g0, d.g1, z, z, z8, 0);
#else
  __builtin_amdgcn_tensor_load_to_lds(g0, d.g1, z, z, 0);
#endif
}

__device__ __forceinline__ unsigned lds_offset_of(const void* p) {
  // LDS flat addresses carry the LDS byte offset in the low 32 bits.
  return (unsigned)(uintptr_t)p;
}

// ---------------------------------------------------------------------------
// f32 -> bf16 linear convert (activations)
// ---------------------------------------------------------------------------
__global__ __launch_bounds__(256)
void cvt_bf16_kernel(const float* __restrict__ in, unsigned short* __restrict__ out, int n) {
  int i = blockIdx.x * blockDim.x + threadIdx.x;
  int stride = gridDim.x * blockDim.x;
  for (; i < n; i += stride) out[i] = f32_to_bf16(in[i]);
}

// ---------------------------------------------------------------------------
// f32 [K][N] -> bf16 transposed [N][K] (weights; paid once)
// ---------------------------------------------------------------------------
__global__ __launch_bounds__(256)
void cvt_transpose_kernel(const float* __restrict__ in, unsigned short* __restrict__ out) {
  __shared__ unsigned short t[32][33];
  const int tx = threadIdx.x & 31, ty = threadIdx.x >> 5;
  const int n0 = blockIdx.x * 32, k0 = blockIdx.y * 32;
#pragma unroll
  for (int i = 0; i < 4; ++i) {
    const int k = k0 + ty + 8 * i;
    t[ty + 8 * i][tx] = f32_to_bf16(in[(size_t)k * DMODEL + n0 + tx]);
  }
  __syncthreads();
#pragma unroll
  for (int i = 0; i < 4; ++i) {
    const int n = n0 + ty + 8 * i;
    out[(size_t)n * DMODEL + k0 + tx] = t[tx][ty + 8 * i];
  }
}

// ---------------------------------------------------------------------------
// Fused QKV projection (N = 3072 = Q|K|V) + RoPE epilogue on Q,K.
// 256 threads = 8 waves; tile 128x128, K-step 32; wave owns 32x64 (8 WMMA/step).
// Double-buffered TDM pipeline: DMA of step k+1 overlaps WMMA of step k.
// ---------------------------------------------------------------------------
__global__ __launch_bounds__(256)
void qkv_rope_wmma(const unsigned short* __restrict__ xb,
                   const unsigned short* __restrict__ wqt,
                   const unsigned short* __restrict__ wkt,
                   const unsigned short* __restrict__ wvt,
                   unsigned short* __restrict__ Qb,
                   unsigned short* __restrict__ Kb,
                   unsigned short* __restrict__ Vb) {
  __shared__ __align__(16) unsigned short As[2][128][40];   // x tile  [m][k], padded
  __shared__ __align__(16) unsigned short Bs[2][128][40];   // W^T tile [n][k], padded

  const int tid = threadIdx.x;
  const int lane = tid & 31, wave = tid >> 5;
  const int lr = lane & 15, g = lane >> 4;
  const int wm = wave & 3, wn = wave >> 2;
  const int n0 = blockIdx.x * 128;       // Q|K|V column space
  const int m0 = blockIdx.y * 128;
  const int matId = n0 >> 10;            // 0=Q 1=K 2=V
  const int nw = n0 & 1023;
  const unsigned short* Wt = (matId == 0) ? wqt : (matId == 1) ? wkt : wvt;
  const unsigned short* Arow = xb + (size_t)m0 * DMODEL;
  const unsigned short* Brow = Wt + (size_t)nw * DMODEL;
  const unsigned ldsA[2] = { lds_offset_of(&As[0][0][0]), lds_offset_of(&As[1][0][0]) };
  const unsigned ldsB[2] = { lds_offset_of(&Bs[0][0][0]), lds_offset_of(&Bs[1][0][0]) };
  const TdmDesc dA = tdm_make_2d(DMODEL, MTOT,   DMODEL, 32, 128, true);
  const TdmDesc dB = tdm_make_2d(DMODEL, DMODEL, DMODEL, 32, 128, true);

  v8f acc[2][4] = {};
  const int NK = DMODEL / 32;

  if (wave == 0) {
    tdm_issue(dA, ldsA[0], Arow);
    tdm_issue(dB, ldsB[0], Brow);
  }
  for (int ks = 0; ks < NK; ++ks) {
    const int cur = ks & 1;
    __syncthreads();                       // readers of buf 'nxt' are done
    if (wave == 0) {
      if (ks + 1 < NK) {
        const int nxt = cur ^ 1;
        tdm_issue(dA, ldsA[nxt], Arow + (ks + 1) * 32);
        tdm_issue(dB, ldsB[nxt], Brow + (ks + 1) * 32);
        __builtin_amdgcn_s_wait_tensorcnt(2);   // in-order: oldest pair (cur) done
      } else {
        __builtin_amdgcn_s_wait_tensorcnt(0);
      }
    }
    __syncthreads();                       // buf[cur] visible to all waves

    Frag16 a[2], b[4];
#pragma unroll
    for (int ms = 0; ms < 2; ++ms) {
      const int mr = wm * 32 + ms * 16 + lr;
      a[ms].q[0] = *(const uint4*)&As[cur][mr][g * 8];
      a[ms].q[1] = *(const uint4*)&As[cur][mr][16 + g * 8];
    }
#pragma unroll
    for (int ns = 0; ns < 4; ++ns) {
      const int nb = wn * 64 + ns * 16 + lr;
      b[ns].q[0] = *(const uint4*)&Bs[cur][nb][g * 8];
      b[ns].q[1] = *(const uint4*)&Bs[cur][nb][16 + g * 8];
    }
#pragma unroll
    for (int ms = 0; ms < 2; ++ms)
#pragma unroll
      for (int ns = 0; ns < 4; ++ns)
        acc[ms][ns] = wmma_bf16(a[ms], b[ns], acc[ms][ns]);
  }

  // ----- epilogue: RoPE (Q,K) + scatter -----
  const int head = (nw + wn * 64) >> 6;            // wave slab = exactly one head
  unsigned short* dst = (matId == 0) ? Qb : (matId == 1) ? Kb : Vb;
  const bool doRope = (matId < 2);
#pragma unroll
  for (int ns = 0; ns < 4; ++ns) {
    const int d = ns * 16 + lr;                    // 0..63 within head
    const float inv = __powf(10000.0f, -(float)(d & ~1) * (1.0f / 64.0f));
#pragma unroll
    for (int ms = 0; ms < 2; ++ms) {
#pragma unroll
      for (int v = 0; v < 8; ++v) {
        float val = acc[ms][ns][v];
        const int m  = m0 + wm * 32 + ms * 16 + v + 8 * g;
        const int s  = m & (SEQ - 1);
        const int bb = m >> 11;
        const float partner = __shfl_xor(val, 1, 32);
        if (doRope) {
          const float ang = (float)s * inv;
          const float cs = __cosf(ang), sn = __sinf(ang);
          val = (d & 1) ? (partner * sn + val * cs)
                        : (val * cs - partner * sn);
        }
        const size_t bh = (size_t)bb * NHEAD + head;
        const size_t idx = (matId == 2) ? (bh * HDIM + d) * SEQ + s   // V^T [b,h,d,s]
                                        : (bh * SEQ + s) * HDIM + d;  // Q,K [b,h,s,d]
        dst[idx] = f32_to_bf16(val);
      }
    }
  }
}

// ---------------------------------------------------------------------------
// Causal flash attention, bf16 WMMA / f32 online softmax.
// 4 waves, 64 query rows; 32-key tiles, double-buffered TDM (K and V^T).
// ---------------------------------------------------------------------------
__global__ __launch_bounds__(128)
void flash_attn_wmma(const unsigned short* __restrict__ Qb,
                     const unsigned short* __restrict__ Kb,
                     const unsigned short* __restrict__ Vb,
                     unsigned short* __restrict__ Ab) {
  __shared__ __align__(16) unsigned short Ks[2][32][64];    // [key][d]
  __shared__ __align__(16) unsigned short Vs[2][64][32];    // [d][key]
  __shared__ __align__(16) unsigned short Ps[4][16][32];    // per-wave P transpose

  const int tid = threadIdx.x, lane = tid & 31, w = tid >> 5;
  const int lr = lane & 15, g = lane >> 4;
  const int qt = blockIdx.x, bh = blockIdx.y;
  const int b = bh >> 4, h = bh & 15;
  const size_t hb = (size_t)bh * SEQ * HDIM;
  const unsigned short* Qh = Qb + hb;
  const unsigned short* Kh = Kb + hb;
  const unsigned short* Vh = Vb + hb;          // V^T: [d][s]
  const unsigned ldsK[2] = { lds_offset_of(&Ks[0][0][0]), lds_offset_of(&Ks[1][0][0]) };
  const unsigned ldsV[2] = { lds_offset_of(&Vs[0][0][0]), lds_offset_of(&Vs[1][0][0]) };
  const TdmDesc dK = tdm_make_2d(HDIM, SEQ, HDIM, 64, 32, false);
  const TdmDesc dV = tdm_make_2d(SEQ, HDIM, SEQ, 32, 64, false);

  // Q A-fragments in registers for the whole kernel (d 0..31 and 32..63)
  Frag16 qf0, qf1;
  {
    const unsigned short* qp = Qh + (size_t)(qt * 64 + w * 16 + lr) * HDIM;
    qf0.q[0] = *(const uint4*)(qp + g * 8);
    qf0.q[1] = *(const uint4*)(qp + 16 + g * 8);
    qf1.q[0] = *(const uint4*)(qp + 32 + g * 8);
    qf1.q[1] = *(const uint4*)(qp + 48 + g * 8);
  }

  v8f O0 = {}, O1 = {}, O2 = {}, O3 = {};
  float mi[8], li[8];
#pragma unroll
  for (int v = 0; v < 8; ++v) { mi[v] = -3.0e38f; li[v] = 0.0f; }

  const int ktiles = 2 * qt + 2;
  if (w == 0) {
    tdm_issue(dK, ldsK[0], Kh);
    tdm_issue(dV, ldsV[0], Vh);
  }
  for (int kt = 0; kt < ktiles; ++kt) {
    const int cur = kt & 1;
    __syncthreads();
    if (w == 0) {
      if (kt + 1 < ktiles) {
        const int nxt = cur ^ 1;
        tdm_issue(dK, ldsK[nxt], Kh + (size_t)((kt + 1) * 32) * HDIM);
        tdm_issue(dV, ldsV[nxt], Vh + (size_t)((kt + 1) * 32));
        __builtin_amdgcn_s_wait_tensorcnt(2);
      } else {
        __builtin_amdgcn_s_wait_tensorcnt(0);
      }
    }
    __syncthreads();

    // S = Q x K^T : two C frags (keys lr, 16+lr), K-dim 64 => 2 WMMA each
    v8f s0 = {}, s1 = {};
    Frag16 kb;
    kb.q[0] = *(const uint4*)&Ks[cur][lr][g * 8];
    kb.q[1] = *(const uint4*)&Ks[cur][lr][16 + g * 8];
    s0 = wmma_bf16(qf0, kb, s0);
    kb.q[0] = *(const uint4*)&Ks[cur][lr][32 + g * 8];
    kb.q[1] = *(const uint4*)&Ks[cur][lr][48 + g * 8];
    s0 = wmma_bf16(qf1, kb, s0);
    kb.q[0] = *(const uint4*)&Ks[cur][16 + lr][g * 8];
    kb.q[1] = *(const uint4*)&Ks[cur][16 + lr][16 + g * 8];
    s1 = wmma_bf16(qf0, kb, s1);
    kb.q[0] = *(const uint4*)&Ks[cur][16 + lr][32 + g * 8];
    kb.q[1] = *(const uint4*)&Ks[cur][16 + lr][48 + g * 8];
    s1 = wmma_bf16(qf1, kb, s1);

    // online softmax (row = v + 8*g lives in one 16-lane half)
    const int key0 = kt * 32 + lr, key1 = key0 + 16;
#pragma unroll
    for (int v = 0; v < 8; ++v) {
      const int q = qt * 64 + w * 16 + v + 8 * g;
      float a = (key0 <= q) ? s0[v] * 0.125f : -3.0e38f;
      float c = (key1 <= q) ? s1[v] * 0.125f : -3.0e38f;
      float mx = fmaxf(a, c);
#pragma unroll
      for (int off = 1; off < 16; off <<= 1) mx = fmaxf(mx, __shfl_xor(mx, off, 32));
      const float mnew = fmaxf(mi[v], mx);
      const float sc = __expf(mi[v] - mnew);
      const float p0 = __expf(a - mnew);
      const float p1 = __expf(c - mnew);
      float rs = p0 + p1;
#pragma unroll
      for (int off = 1; off < 16; off <<= 1) rs += __shfl_xor(rs, off, 32);
      li[v] = li[v] * sc + rs;
      mi[v] = mnew;
      O0[v] *= sc; O1[v] *= sc; O2[v] *= sc; O3[v] *= sc;
      Ps[w][v + 8 * g][lr]      = f32_to_bf16(p0);
      Ps[w][v + 8 * g][16 + lr] = f32_to_bf16(p1);
    }

    // O += P x V (P re-read in A layout from wave-private LDS)
    Frag16 pf, vf;
    pf.q[0] = *(const uint4*)&Ps[w][lr][g * 8];
    pf.q[1] = *(const uint4*)&Ps[w][lr][16 + g * 8];
    vf.q[0] = *(const uint4*)&Vs[cur][lr][g * 8];
    vf.q[1] = *(const uint4*)&Vs[cur][lr][16 + g * 8];
    O0 = wmma_bf16(pf, vf, O0);
    vf.q[0] = *(const uint4*)&Vs[cur][16 + lr][g * 8];
    vf.q[1] = *(const uint4*)&Vs[cur][16 + lr][16 + g * 8];
    O1 = wmma_bf16(pf, vf, O1);
    vf.q[0] = *(const uint4*)&Vs[cur][32 + lr][g * 8];
    vf.q[1] = *(const uint4*)&Vs[cur][32 + lr][16 + g * 8];
    O2 = wmma_bf16(pf, vf, O2);
    vf.q[0] = *(const uint4*)&Vs[cur][48 + lr][g * 8];
    vf.q[1] = *(const uint4*)&Vs[cur][48 + lr][16 + g * 8];
    O3 = wmma_bf16(pf, vf, O3);
  }

#pragma unroll
  for (int v = 0; v < 8; ++v) {
    const int q = qt * 64 + w * 16 + v + 8 * g;
    const float inv = 1.0f / li[v];
    const size_t base = ((size_t)b * SEQ + q) * DMODEL + h * HDIM + lr;
    Ab[base]      = f32_to_bf16(O0[v] * inv);
    Ab[base + 16] = f32_to_bf16(O1[v] * inv);
    Ab[base + 32] = f32_to_bf16(O2[v] * inv);
    Ab[base + 48] = f32_to_bf16(O3[v] * inv);
  }
}

// ---------------------------------------------------------------------------
// Output projection: out(f32) = Ab(bf16) x Wo^T(bf16 [N][K]) + bias
// Same double-buffered TDM-fed 128x128 tiling as QKV.
// ---------------------------------------------------------------------------
__global__ __launch_bounds__(256)
void outproj_wmma(const unsigned short* __restrict__ Ab,
                  const unsigned short* __restrict__ wot,
                  const float* __restrict__ bias,
                  float* __restrict__ out) {
  __shared__ __align__(16) unsigned short As[2][128][40];
  __shared__ __align__(16) unsigned short Bs[2][128][40];
  const int tid = threadIdx.x, lane = tid & 31, wave = tid >> 5;
  const int lr = lane & 15, g = lane >> 4;
  const int wm = wave & 3, wn = wave >> 2;
  const int n0 = blockIdx.x * 128, m0 = blockIdx.y * 128;
  const unsigned short* Arow = Ab + (size_t)m0 * DMODEL;
  const unsigned short* Brow = wot + (size_t)n0 * DMODEL;
  const unsigned ldsA[2] = { lds_offset_of(&As[0][0][0]), lds_offset_of(&As[1][0][0]) };
  const unsigned ldsB[2] = { lds_offset_of(&Bs[0][0][0]), lds_offset_of(&Bs[1][0][0]) };
  const TdmDesc dA = tdm_make_2d(DMODEL, MTOT,   DMODEL, 32, 128, true);
  const TdmDesc dB = tdm_make_2d(DMODEL, DMODEL, DMODEL, 32, 128, true);

  v8f acc[2][4] = {};
  const int NK = DMODEL / 32;

  if (wave == 0) {
    tdm_issue(dA, ldsA[0], Arow);
    tdm_issue(dB, ldsB[0], Brow);
  }
  for (int ks = 0; ks < NK; ++ks) {
    const int cur = ks & 1;
    __syncthreads();
    if (wave == 0) {
      if (ks + 1 < NK) {
        const int nxt = cur ^ 1;
        tdm_issue(dA, ldsA[nxt], Arow + (ks + 1) * 32);
        tdm_issue(dB, ldsB[nxt], Brow + (ks + 1) * 32);
        __builtin_amdgcn_s_wait_tensorcnt(2);
      } else {
        __builtin_amdgcn_s_wait_tensorcnt(0);
      }
    }
    __syncthreads();

    Frag16 a[2], b[4];
#pragma unroll
    for (int ms = 0; ms < 2; ++ms) {
      const int mr = wm * 32 + ms * 16 + lr;
      a[ms].q[0] = *(const uint4*)&As[cur][mr][g * 8];
      a[ms].q[1] = *(const uint4*)&As[cur][mr][16 + g * 8];
    }
#pragma unroll
    for (int ns = 0; ns < 4; ++ns) {
      const int nb = wn * 64 + ns * 16 + lr;
      b[ns].q[0] = *(const uint4*)&Bs[cur][nb][g * 8];
      b[ns].q[1] = *(const uint4*)&Bs[cur][nb][16 + g * 8];
    }
#pragma unroll
    for (int ms = 0; ms < 2; ++ms)
#pragma unroll
      for (int ns = 0; ns < 4; ++ns)
        acc[ms][ns] = wmma_bf16(a[ms], b[ns], acc[ms][ns]);
  }

#pragma unroll
  for (int ns = 0; ns < 4; ++ns) {
    const int n = n0 + wn * 64 + ns * 16 + lr;
    const float bv = bias[n];
#pragma unroll
    for (int ms = 0; ms < 2; ++ms) {
#pragma unroll
      for (int v = 0; v < 8; ++v) {
        const int m = m0 + wm * 32 + ms * 16 + v + 8 * g;
        out[(size_t)m * DMODEL + n] = acc[ms][ns][v] + bv;
      }
    }
  }
}

// ---------------------------------------------------------------------------
extern "C" void kernel_launch(void* const* d_in, const int* in_sizes, int n_in,
                              void* d_out, int out_size, void* d_ws, size_t ws_size,
                              hipStream_t stream) {
  const float* x    = (const float*)d_in[0];
  const float* wq   = (const float*)d_in[1];
  const float* wk   = (const float*)d_in[2];
  const float* wv   = (const float*)d_in[3];
  const float* wo   = (const float*)d_in[4];
  const float* wo_b = (const float*)d_in[5];
  float* out = (float*)d_out;

  char* ws = (char*)d_ws;
  size_t off = 0;
  const size_t XB  = (size_t)MTOT * DMODEL;               // 4M elems
  const size_t WB  = (size_t)DMODEL * DMODEL;             // 1M elems
  const size_t QKB = (size_t)BATCH * NHEAD * SEQ * HDIM;  // 4M elems
  unsigned short* xb  = (unsigned short*)(ws + off); off += XB * 2;
  unsigned short* wqt = (unsigned short*)(ws + off); off += WB * 2;
  unsigned short* wkt = (unsigned short*)(ws + off); off += WB * 2;
  unsigned short* wvt = (unsigned short*)(ws + off); off += WB * 2;
  unsigned short* wot = (unsigned short*)(ws + off); off += WB * 2;
  unsigned short* Qb  = (unsigned short*)(ws + off); off += QKB * 2;
  unsigned short* Kb  = (unsigned short*)(ws + off); off += QKB * 2;
  unsigned short* Vb  = (unsigned short*)(ws + off); off += QKB * 2;   // [b,h,d,s]
  unsigned short* Abf = (unsigned short*)(ws + off); off += XB * 2;    // ~48 MiB total

  cvt_bf16_kernel<<<2048, 256, 0, stream>>>(x, xb, (int)XB);
  cvt_transpose_kernel<<<dim3(32, 32), 256, 0, stream>>>(wq, wqt);
  cvt_transpose_kernel<<<dim3(32, 32), 256, 0, stream>>>(wk, wkt);
  cvt_transpose_kernel<<<dim3(32, 32), 256, 0, stream>>>(wv, wvt);
  cvt_transpose_kernel<<<dim3(32, 32), 256, 0, stream>>>(wo, wot);

  qkv_rope_wmma<<<dim3(NQKV / 128, MTOT / 128), 256, 0, stream>>>(
      xb, wqt, wkt, wvt, Qb, Kb, Vb);

  flash_attn_wmma<<<dim3(SEQ / 64, BATCH * NHEAD), 128, 0, stream>>>(
      Qb, Kb, Vb, Abf);

  outproj_wmma<<<dim3(DMODEL / 128, MTOT / 128), 256, 0, stream>>>(
      Abf, wot, wo_b, out);
}
